// Net_44787918963235
// MI455X (gfx1250) — matrix operand order repaired and checked by reading, not compiled, not run on hardware
//
#include <hip/hip_runtime.h>
#include <math.h>

// ---------------------------------------------------------------------------
// Pan-Mamba pansharpening network for MI455X (gfx1250, wave32, WMMA)
//  * convs: LDS im2col ([x][k] layout -> contiguous ds_load_b128 fragments)
//  * mamba projections: WMMA GEMM, one wave per 16-row strip covering all N
//    tiles (A fragment loaded once per K chunk), weights pre-packed to f16
//  * selective scan: VALU, LDS-staged B/C
//  * all epilogue variants (leaky / residual / add) are template parameters
//    so the unrolled store loops are branch-free straight-line code
// ---------------------------------------------------------------------------

typedef __attribute__((ext_vector_type(16))) _Float16 v16h;
typedef __attribute__((ext_vector_type(8)))  _Float16 v8h;
typedef __attribute__((ext_vector_type(8)))  float    v8f;

#define WMMA_F16(a, b, c) \
  __builtin_amdgcn_wmma_f32_16x16x32_f16(false, (a), false, (b), (short)0, (c), false, false)

__device__ __forceinline__ float leaky_f(float v)    { return v >= 0.f ? v : 0.2f * v; }
__device__ __forceinline__ float silu_f(float v)     { return v / (1.f + __expf(-v)); }
__device__ __forceinline__ float softplus_f(float v) { return (v > 20.f) ? v : log1pf(__expf(v)); }

// --------------------------- nearest-x4 upsample ---------------------------
__global__ void upsample4_kernel(const float* __restrict__ ms, float* __restrict__ out) {
  int i = blockIdx.x * blockDim.x + threadIdx.x;        // 4*4*64*64
  if (i >= 65536) return;
  int x = i & 63, y = (i >> 6) & 63, c = (i >> 12) & 3, n = i >> 14;
  out[i] = ms[((n * 4 + c) << 8) + ((y >> 2) << 4) + (x >> 2)];
}

// ----------------------- f32 -> packed f16 convert -------------------------
__global__ void f32_to_f16_kernel(const float* __restrict__ src, _Float16* __restrict__ dst, int n) {
  int i = blockIdx.x * blockDim.x + threadIdx.x;
  if (i < n) dst[i] = (_Float16)src[i];
}

// ------------------- 3x3 conv, pad 1, as LDS-im2col WMMA GEMM --------------
// in: (4,CIN,64,64)  w: (32,CIN,3,3)  out: (4,32,64,64)
// block = 128 threads (4 waves) = one scanline (64 px) x 32 out channels.
template <int CIN, bool LEAKY, bool RES>
__global__ void conv3x3_wmma_kernel(const float* __restrict__ in,
                                    const float* __restrict__ w,
                                    const float* __restrict__ bias,
                                    const float* __restrict__ res,
                                    float* __restrict__ out) {
  constexpr int K    = CIN * 9;
  constexpr int Kp   = (K + 31) & ~31;
  constexpr int AROW = Kp + 8;              // pad to stagger LDS banks
  __shared__ _Float16 sA[64 * AROW];        // im2col tile: [x][k]
  __shared__ _Float16 wl[32 * Kp];          // weights:     [co][k]
  const int tid  = threadIdx.x;
  const int bimg = blockIdx.x >> 6;
  const int y    = blockIdx.x & 63;

  for (int idx = tid; idx < 32 * Kp; idx += 128) {      // stage weights
    int co = idx / Kp, k = idx % Kp;
    wl[idx] = (_Float16)((k < K) ? w[co * K + k] : 0.f);
  }
  for (int idx = tid; idx < 64 * Kp; idx += 128) {      // stage im2col (zero-padded)
    int x = idx / Kp, k = idx % Kp;
    float v = 0.f;
    if (k < K) {
      int ci = k / 9, r = k % 9, kh = r / 3, kw = r % 3;
      int yy = y + kh - 1, xx = x + kw - 1;
      if (yy >= 0 && yy < 64 && xx >= 0 && xx < 64)
        v = in[((bimg * CIN + ci) << 12) + (yy << 6) + xx];
    }
    sA[x * AROW + k] = (_Float16)v;
  }
  __syncthreads();

  const int lane = tid & 31, wv = tid >> 5;
  const int half_sel = lane >> 4, lrow = lane & 15;
  const int x0 = wv * 16;
  const int xa = x0 + lrow;                 // this lane's A-row pixel

  v8f acc0 = {}, acc1 = {};
#pragma unroll
  for (int kk = 0; kk < Kp; kk += 32) {
    const int ka = kk + (half_sel << 4);    // lane's K base for A (contiguous 16)
    v8h lo = *(const v8h*)&sA[xa * AROW + ka];        // k = ka .. ka+7
    v8h hi = *(const v8h*)&sA[xa * AROW + ka + 8];    // k = ka+8 .. ka+15
    v16h a = __builtin_shufflevector(lo, hi, 0, 1, 2, 3, 4, 5, 6, 7,
                                     8, 9, 10, 11, 12, 13, 14, 15);
    const int kb = kk + (half_sel << 4);    // lane's K base for B (contiguous 16)
    v16h b0 = *(const v16h*)&wl[lrow * Kp + kb];
    v16h b1 = *(const v16h*)&wl[(16 + lrow) * Kp + kb];
    acc0 = WMMA_F16(a, b0, acc0);
    acc1 = WMMA_F16(a, b1, acc1);
  }
#pragma unroll
  for (int r = 0; r < 8; ++r) {             // D layout epilogue (branch-free)
    int p   = (y << 6) + x0 + half_sel * 8 + r;
    int co0 = lrow, co1 = 16 + lrow;
    float v0 = acc0[r] + bias[co0];
    float v1 = acc1[r] + bias[co1];
    if (LEAKY) { v0 = leaky_f(v0); v1 = leaky_f(v1); }
    if (RES) {
      v0 += res[((bimg * 32 + co0) << 12) + p];
      v1 += res[((bimg * 32 + co1) << 12) + p];
    }
    out[((bimg * 32 + co0) << 12) + p] = v0;
    out[((bimg * 32 + co1) << 12) + p] = v1;
  }
}

// -------------------- instance-norm on channels 0..15 ----------------------
__global__ void inorm16_kernel(float* __restrict__ x, const float* __restrict__ w,
                               const float* __restrict__ b) {
  const int n = blockIdx.x >> 4, c = blockIdx.x & 15;   // 64 blocks
  __shared__ float ssum[256], ssq[256];
  float* base = x + ((n * 32 + c) << 12);
  float s = 0.f, q = 0.f;
  for (int i = threadIdx.x; i < 4096; i += 256) { float v = base[i]; s += v; q += v * v; }
  ssum[threadIdx.x] = s; ssq[threadIdx.x] = q;
  __syncthreads();
  for (int st = 128; st > 0; st >>= 1) {
    if (threadIdx.x < st) { ssum[threadIdx.x] += ssum[threadIdx.x + st];
                            ssq[threadIdx.x]  += ssq[threadIdx.x + st]; }
    __syncthreads();
  }
  float mu  = ssum[0] * (1.f / 4096.f);
  float var = ssq[0] * (1.f / 4096.f) - mu * mu;
  float inv = rsqrtf(var + 1e-5f);
  float ww = w[c], bb = b[c];
  for (int i = threadIdx.x; i < 4096; i += 256)
    base[i] = (base[i] - mu) * inv * ww + bb;
}

// --------------------------- strip WMMA GEMM -------------------------------
// out[M][N] = A * Wh^T + bias.  One wave per 16-row strip; the A fragment is
// loaded once per K chunk and reused across ALL N tiles (kept in VGPRs).
// Wh is pre-packed f16 [N][K].  AL=0: A row-major [M][K] (float4 loads);
// AL=1: A is NCHW feature (4,K,64,64) for the 1x1-conv tokenizer.
template <int K, int N, int AL, bool BIAS>
__global__ void wmma_gemm_kernel(const float* __restrict__ A, const _Float16* __restrict__ Wh,
                                 const float* __restrict__ bias, float* __restrict__ out) {
  constexpr int NT = (N + 15) / 16;
  const int tm = blockIdx.x;                // 1024 strips of 16 rows
  const int lane = threadIdx.x & 31;
  const int half_sel = lane >> 4, lrow = lane & 15;
  const int arow = tm * 16 + lrow;

  v8f acc[NT];
#pragma unroll
  for (int t = 0; t < NT; ++t) acc[t] = (v8f){};

#pragma unroll
  for (int kk = 0; kk < K; kk += 32) {
    v16h a;
    if (AL == 0) {
      const float* ap = A + (size_t)arow * K + kk + (half_sel << 4);
      const float4 q0 = *(const float4*)(ap + 0);
      const float4 q1 = *(const float4*)(ap + 4);
      const float4 q2 = *(const float4*)(ap + 8);
      const float4 q3 = *(const float4*)(ap + 12);
      a[0]  = (_Float16)q0.x; a[1]  = (_Float16)q0.y; a[2]  = (_Float16)q0.z; a[3]  = (_Float16)q0.w;
      a[4]  = (_Float16)q1.x; a[5]  = (_Float16)q1.y; a[6]  = (_Float16)q1.z; a[7]  = (_Float16)q1.w;
      a[8]  = (_Float16)q2.x; a[9]  = (_Float16)q2.y; a[10] = (_Float16)q2.z; a[11] = (_Float16)q2.w;
      a[12] = (_Float16)q3.x; a[13] = (_Float16)q3.y; a[14] = (_Float16)q3.z; a[15] = (_Float16)q3.w;
    } else {
      const int bi = arow >> 12, p = arow & 4095;
      const float* ap = A + ((size_t)bi * K << 12) + p;
#pragma unroll
      for (int h = 0; h < 16; ++h) {
        int k = kk + (half_sel << 4) + h;   // k < K (K=32)
        a[h] = (_Float16)ap[(size_t)k << 12];
      }
    }
    const int kb = kk + (half_sel << 4);
#pragma unroll
    for (int t = 0; t < NT; ++t) {
      const int col = t * 16 + lrow;
      v16h b;
      if (col < N) {                        // compile-time true except last tile
        b = *(const v16h*)&Wh[(size_t)col * K + kb];
      } else {
#pragma unroll
        for (int h = 0; h < 16; ++h) b[h] = (_Float16)0.f;
      }
      acc[t] = WMMA_F16(a, b, acc[t]);
    }
  }
#pragma unroll
  for (int t = 0; t < NT; ++t) {
    const int col = t * 16 + lrow;
    if (col < N) {
      float bv = BIAS ? bias[col] : 0.f;
#pragma unroll
      for (int r = 0; r < 8; ++r) {
        int row = tm * 16 + half_sel * 8 + r;
        out[(size_t)row * N + col] = acc[t][r] + bv;
      }
    }
  }
}

// ---------------- residual add + double LayerNorm (C = 32) -----------------
template <bool ADD>
__global__ void add_ln2_kernel(const float* __restrict__ a, const float* __restrict__ badd,
                               float* __restrict__ resi_out, const float* __restrict__ w,
                               const float* __restrict__ bias, float* __restrict__ out) {
  int t = blockIdx.x * blockDim.x + threadIdx.x;        // 16384 tokens
  if (t >= 16384) return;
  float v[32];
  const float* pa = a + (size_t)t * 32;
#pragma unroll
  for (int c = 0; c < 32; ++c) {
    float x = pa[c];
    if (ADD) x += badd[(size_t)t * 32 + c];
    v[c] = x;
    if (ADD) resi_out[(size_t)t * 32 + c] = x;
  }
#pragma unroll
  for (int pass = 0; pass < 2; ++pass) {                // norm applied TWICE
    float mu = 0.f;
#pragma unroll
    for (int c = 0; c < 32; ++c) mu += v[c];
    mu *= (1.f / 32.f);
    float var = 0.f;
#pragma unroll
    for (int c = 0; c < 32; ++c) { float d = v[c] - mu; var += d * d; }
    var *= (1.f / 32.f);
    float inv = rsqrtf(var + 1e-5f);
#pragma unroll
    for (int c = 0; c < 32; ++c) v[c] = (v[c] - mu) * inv * w[c] + bias[c];
  }
#pragma unroll
  for (int c = 0; c < 32; ++c) out[(size_t)t * 32 + c] = v[c];
}

// ---------------- depthwise conv1d (k=4, pad 1,2) + SiLU -------------------
__global__ void dwconv1d_silu_kernel(const float* __restrict__ xz, const float* __restrict__ w,
                                     const float* __restrict__ bias, float* __restrict__ u) {
  int i = blockIdx.x * blockDim.x + threadIdx.x;        // 4*4096*64
  if (i >= 1048576) return;
  int d = i & 63, l = (i >> 6) & 4095, bn = i >> 18;
  const float* xb = xz + (size_t)(bn * 4096) * 128 + d; // x = xz[..., 0:64]
  float s = bias[d];
#pragma unroll
  for (int j = 0; j < 4; ++j) {
    int ll = l + j - 1;
    if (ll >= 0 && ll < 4096) s += w[d * 4 + j] * xb[(size_t)ll * 128];
  }
  u[(size_t)(bn * 4096 + l) * 64 + d] = silu_f(s);
}

// ------------------------- delta = softplus(dt proj) -----------------------
__global__ void delta_kernel(const float* __restrict__ dbl, const float* __restrict__ dt_w,
                             const float* __restrict__ dt_b, float* __restrict__ delta) {
  int i = blockIdx.x * blockDim.x + threadIdx.x;        // 16384*64
  if (i >= 1048576) return;
  int d = i & 63; int t = i >> 6;
  const float* row = dbl + (size_t)t * 34;
  float s = dt_b[d] + row[0] * dt_w[d * 2 + 0] + row[1] * dt_w[d * 2 + 1];
  delta[i] = softplus_f(s);
}

// --------------------------- selective scan --------------------------------
__global__ void sel_scan_kernel(const float* __restrict__ u, const float* __restrict__ delta,
                                const float* __restrict__ dbl, const float* __restrict__ A_log,
                                const float* __restrict__ D, float* __restrict__ y) {
  const int bn = blockIdx.x, d = threadIdx.x;           // 4 blocks x 64 threads
  __shared__ float sB[16], sC[16];
  float A[16], h[16];
#pragma unroll
  for (int n = 0; n < 16; ++n) { A[n] = -__expf(A_log[d * 16 + n]); h[n] = 0.f; }
  const float Dd = D[d];
  for (int l = 0; l < 4096; ++l) {
    const int t = bn * 4096 + l;
    if (d < 16)      sB[d]      = dbl[(size_t)t * 34 + 2 + d];
    else if (d < 32) sC[d - 16] = dbl[(size_t)t * 34 + 18 + (d - 16)];
    __syncthreads();
    float del = delta[(size_t)t * 64 + d];
    float uu  = u[(size_t)t * 64 + d];
    float du  = del * uu;
    float acc = 0.f;
#pragma unroll
    for (int n = 0; n < 16; ++n) {
      h[n] = __expf(del * A[n]) * h[n] + du * sB[n];
      acc += h[n] * sC[n];
    }
    y[(size_t)t * 64 + d] = acc + uu * Dd;
    __syncthreads();
  }
}

// -------------------------- cross gating combine ---------------------------
__global__ void gate_kernel(const float* __restrict__ ya, const float* __restrict__ yb,
                            const float* __restrict__ xza, const float* __restrict__ xzb,
                            float* __restrict__ out) {
  int i = blockIdx.x * blockDim.x + threadIdx.x;        // 16384*64
  if (i >= 1048576) return;
  int d = i & 63; int t = i >> 6;
  float za = xza[(size_t)t * 128 + 64 + d];
  float zb = xzb[(size_t)t * 128 + 64 + d];
  out[i] = ya[i] * silu_f(zb) + yb[i] * silu_f(za);
}

// ------------------ depthwise 3x3 on tokens (b,l,32) -----------------------
__global__ void dw3x3_tokens_kernel(const float* __restrict__ g, const float* __restrict__ w,
                                    const float* __restrict__ b, float* __restrict__ out) {
  int i = blockIdx.x * blockDim.x + threadIdx.x;        // 4*32*64*64
  if (i >= 524288) return;
  int x = i & 63, y = (i >> 6) & 63, c = (i >> 12) & 31, bn = i >> 17;
  float s = b[c];
#pragma unroll
  for (int dy = 0; dy < 3; ++dy)
#pragma unroll
    for (int dx = 0; dx < 3; ++dx) {
      int yy = y + dy - 1, xx = x + dx - 1;
      if (yy >= 0 && yy < 64 && xx >= 0 && xx < 64)
        s += w[c * 9 + dy * 3 + dx] * g[(size_t)(bn * 4096 + yy * 64 + xx) * 32 + c];
    }
  out[(size_t)(bn * 4096 + y * 64 + x) * 32 + c] = s;
}

__global__ void tokens_to_nchw_kernel(const float* __restrict__ t, float* __restrict__ out) {
  int i = blockIdx.x * blockDim.x + threadIdx.x;        // 4*32*4096
  if (i >= 524288) return;
  int p = i & 4095, c = (i >> 12) & 31, bn = i >> 17;
  out[i] = t[(size_t)(bn * 4096 + p) * 32 + c];
}

__global__ void fill_zero_kernel(float* __restrict__ p, int n) {
  int i = blockIdx.x * blockDim.x + threadIdx.x;
  if (i < n) p[i] = 0.f;
}

// ===========================================================================
extern "C" void kernel_launch(void* const* d_in, const int* in_sizes, int n_in,
                              void* d_out, int out_size, void* d_ws, size_t ws_size,
                              hipStream_t stream) {
  (void)in_sizes; (void)n_in; (void)out_size; (void)ws_size;
  auto in = [&](int i) { return (const float*)d_in[i]; };

  float* ws = (float*)d_ws;
  size_t off = 0;
  auto alloc  = [&](size_t n) { float* p = ws + off; off += n; return p; };
  auto halloc = [&](size_t n) { return (_Float16*)alloc((n + 1) / 2); };
  const int F = 524288;                 // 4*32*64*64 tokens / feature map
  float* ms_bic = alloc(65536);
  float* fA   = alloc(F); float* fB = alloc(F); float* fC = alloc(F);
  float* ms_t = alloc(F); float* pan_t = alloc(F); float* ms_t2 = alloc(F);
  float* resi = alloc(F);
  float* msn2 = alloc(F); float* pann2 = alloc(F);
  float* xz_a = alloc(2097152); float* xz_b = alloc(2097152);
  float* u_a  = alloc(1048576); float* u_b  = alloc(1048576);
  float* dbl_a = alloc(557056); float* dbl_b = alloc(557056);
  float* del_a = alloc(1048576); float* del_b = alloc(1048576);
  float* y_a  = alloc(1048576); float* y_b  = alloc(1048576);
  float* ycmb = alloc(1048576); float* gbuf = alloc(F);

  // ---- pre-pack all GEMM weights to f16 (static inputs, done every call) --
  auto cvt = [&](const float* src, _Float16* dst, int n) {
    f32_to_f16_kernel<<<(n + 255) / 256, 256, 0, stream>>>(src, dst, n);
  };
  _Float16* tok_ms  = halloc(1024);  cvt(in(42), tok_ms, 1024);
  _Float16* tok_pan = halloc(1024);  cvt(in(44), tok_pan, 1024);
  _Float16 *h_in_a[5], *h_in_b[5], *h_xp_a[5], *h_xp_b[5], *h_out[5];
  for (int f = 0; f < 5; ++f) {
    const int fb = 46 + f * 23;
    h_in_a[f] = halloc(4096); cvt(in(fb + 6),  h_in_a[f], 4096);
    h_in_b[f] = halloc(4096); cvt(in(fb + 7),  h_in_b[f], 4096);
    h_xp_a[f] = halloc(2176); cvt(in(fb + 12), h_xp_a[f], 2176);
    h_xp_b[f] = halloc(2176); cvt(in(fb + 13), h_xp_b[f], 2176);
    h_out[f]  = halloc(2048); cvt(in(fb + 22), h_out[f],  2048);
  }

  // ---- encoders -----------------------------------------------------------
  upsample4_kernel<<<256, 256, 0, stream>>>(in(0), ms_bic);

  // ms encoder (Cin=4), params[2..21]
  conv3x3_wmma_kernel<4, false, false><<<256, 128, 0, stream>>>(ms_bic, in(2), in(3), nullptr, fA);
  {
    const float* cur = fA; float* nxt = fB;
    for (int blk = 0; blk < 3; ++blk) {
      int q = 4 + blk * 6;
      conv3x3_wmma_kernel<32, true, false><<<256, 128, 0, stream>>>(cur, in(q + 0), in(q + 1), nullptr, fC);
      inorm16_kernel<<<64, 256, 0, stream>>>(fC, in(q + 4), in(q + 5));
      conv3x3_wmma_kernel<32, true, true><<<256, 128, 0, stream>>>(fC, in(q + 2), in(q + 3), cur, nxt);
      const float* t = cur; cur = nxt; nxt = (float*)t;
    }
    wmma_gemm_kernel<32, 32, 1, true><<<1024, 32, 0, stream>>>(cur, tok_ms, in(43), ms_t);
  }
  // pan encoder (Cin=1), params[22..41]
  conv3x3_wmma_kernel<1, false, false><<<256, 128, 0, stream>>>(in(1), in(22), in(23), nullptr, fA);
  {
    const float* cur = fA; float* nxt = fB;
    for (int blk = 0; blk < 3; ++blk) {
      int q = 24 + blk * 6;
      conv3x3_wmma_kernel<32, true, false><<<256, 128, 0, stream>>>(cur, in(q + 0), in(q + 1), nullptr, fC);
      inorm16_kernel<<<64, 256, 0, stream>>>(fC, in(q + 4), in(q + 5));
      conv3x3_wmma_kernel<32, true, true><<<256, 128, 0, stream>>>(fC, in(q + 2), in(q + 3), cur, nxt);
      const float* t = cur; cur = nxt; nxt = (float*)t;
    }
    wmma_gemm_kernel<32, 32, 1, true><<<1024, 32, 0, stream>>>(cur, tok_pan, in(45), pan_t);
  }

  fill_zero_kernel<<<2048, 256, 0, stream>>>(resi, F);

  // ---- 5 fusion blocks ----------------------------------------------------
  float* cur_ms = ms_t; float* nxt_ms = ms_t2;
  for (int f = 0; f < 5; ++f) {
    const int fb = 46 + f * 23;
    add_ln2_kernel<true><<<64, 256, 0, stream>>>(cur_ms, resi, resi, in(fb), in(fb + 1), msn2);
    add_ln2_kernel<false><<<64, 256, 0, stream>>>(pan_t, nullptr, nullptr, in(fb + 2), in(fb + 3), pann2);
    // in_proj (32 -> 128), no bias
    wmma_gemm_kernel<32, 128, 0, false><<<1024, 32, 0, stream>>>(msn2, h_in_a[f], nullptr, xz_a);
    wmma_gemm_kernel<32, 128, 0, false><<<1024, 32, 0, stream>>>(pann2, h_in_b[f], nullptr, xz_b);
    // depthwise conv1d + SiLU
    dwconv1d_silu_kernel<<<4096, 256, 0, stream>>>(xz_a, in(fb + 8), in(fb + 9), u_a);
    dwconv1d_silu_kernel<<<4096, 256, 0, stream>>>(xz_b, in(fb + 10), in(fb + 11), u_b);
    // x_proj (64 -> 34)
    wmma_gemm_kernel<64, 34, 0, false><<<1024, 32, 0, stream>>>(u_a, h_xp_a[f], nullptr, dbl_a);
    wmma_gemm_kernel<64, 34, 0, false><<<1024, 32, 0, stream>>>(u_b, h_xp_b[f], nullptr, dbl_b);
    // delta = softplus(dt @ dt_w.T + dt_bias)
    delta_kernel<<<4096, 256, 0, stream>>>(dbl_a, in(fb + 14), in(fb + 15), del_a);
    delta_kernel<<<4096, 256, 0, stream>>>(dbl_b, in(fb + 16), in(fb + 17), del_b);
    // selective scan
    sel_scan_kernel<<<4, 64, 0, stream>>>(u_a, del_a, dbl_a, in(fb + 18), in(fb + 20), y_a);
    sel_scan_kernel<<<4, 64, 0, stream>>>(u_b, del_b, dbl_b, in(fb + 19), in(fb + 21), y_b);
    // cross gating + shared out_proj (64 -> 32)
    gate_kernel<<<4096, 256, 0, stream>>>(y_a, y_b, xz_a, xz_b, ycmb);
    wmma_gemm_kernel<64, 32, 0, false><<<1024, 32, 0, stream>>>(ycmb, h_out[f], nullptr, gbuf);
    // depthwise 3x3 back onto tokens
    dw3x3_tokens_kernel<<<2048, 256, 0, stream>>>(gbuf, in(fb + 4), in(fb + 5), nxt_ms);
    float* t = cur_ms; cur_ms = nxt_ms; nxt_ms = t;
  }

  tokens_to_nchw_kernel<<<2048, 256, 0, stream>>>(cur_ms, (float*)d_out);
}
